// MixedLayerCond_36859409334836
// MI455X (gfx1250) — matrix operand order repaired and checked by reading, not compiled, not run on hardware
//
#include <hip/hip_runtime.h>

typedef __attribute__((ext_vector_type(16))) _Float16 v16h;
typedef __attribute__((ext_vector_type(8)))  _Float16 v8h;
typedef __attribute__((ext_vector_type(4)))  _Float16 v4h;
typedef __attribute__((ext_vector_type(2)))  _Float16 v2h;
typedef __attribute__((ext_vector_type(8)))  float    v8f;
typedef __attribute__((ext_vector_type(4)))  float    v4f;

union Frag { v16h v; v8h h[2]; };

#define IN_C  256
#define OUT_C 256
#define NPIX  1024      // 32x32
#define KP    72        // padded LDS row pitch in f16 (16B-aligned, conflict-free)
#define PW38  38        // padded image width/height (border 3 each side)
#define PIMG  1444      // 38*38

// ---------------- prep kernels (run once per launch into d_ws) ----------------

// x[64][256][32][32] f32 -> xpad[64][256][38][38] f16, zero borders of 3
__global__ __launch_bounds__(256)
void pad_x_f16(const float* __restrict__ x, _Float16* __restrict__ xpad) {
    const int bc = blockIdx.x;                       // b*256 + c
    const float* src = x + (size_t)bc * NPIX;
    _Float16*    dst = xpad + (size_t)bc * PIMG;
    for (int e = threadIdx.x; e < PIMG; e += 256) {
        int r  = e / PW38, col = e - r * PW38;
        int hr = r - 3,    hc  = col - 3;
        float v = ((unsigned)hr < 32u && (unsigned)hc < 32u) ? src[hr * 32 + hc] : 0.f;
        dst[e] = (_Float16)v;
    }
}

// w[o][c][kh][kw] f32 -> wh[o][kh*K+kw][c] f16   (K reordered to (kh,kw,c))
template<int K>
__global__ __launch_bounds__(256)
void conv_w_f16(const float* __restrict__ w, _Float16* __restrict__ wh) {
    constexpr int KK = K * K, Ktot = IN_C * KK;
    const int o = blockIdx.x;
    const float* src = w  + (size_t)o * Ktot;
    _Float16*    dst = wh + (size_t)o * Ktot;
    for (int i = threadIdx.x; i < Ktot; i += 256) {
        int r = i >> 8;              // (kh,kw) plane
        int c = i & 255;             // channel
        dst[i] = (_Float16)src[c * KK + r];
    }
}

// ---------------- main conv kernel ----------------
// Block: 256 threads = 8 wave32. Tile: 128 (O) x 128 (pixels), K_TILE = 64.
template<int K, int ARC, bool PREP>
__global__ __launch_bounds__(256)
void cond_conv_wmma(const float* __restrict__ x, const int* __restrict__ y,
                    const int* __restrict__ sample_arc,
                    const float* __restrict__ w, const float* __restrict__ e,
                    const _Float16* __restrict__ xpad,   // PREP only
                    const _Float16* __restrict__ wh,     // PREP only
                    float* __restrict__ out)
{
    const int bx = blockIdx.x;
    const int b  = bx >> 4;
    if (sample_arc[b] != ARC) return;             // uniform exit; EXEC stays all-1s
    const int rem    = bx & 15;
    const int m_base = (rem >> 3) * 128;
    const int n_base = (rem & 7)  * 128;

    constexpr int KK   = K * K;
    constexpr int Ktot = IN_C * KK;               // divisible by 64 for all K
    constexpr int PAD  = K / 2;

    __shared__ alignas(16) _Float16 ldsA[128 * KP];   // weights [m][kk]
    __shared__ alignas(16) _Float16 ldsB[128 * KP];   // im2col  [n][kk]
    __shared__ float biasLds[128];

    const int t = threadIdx.x;
    float* outb = out + (size_t)b * OUT_C * NPIX;

    if (t < 128) biasLds[t] = e[(size_t)y[b] * OUT_C + m_base + t];

    const int lane = t & 31;
    const int wave = t >> 5;
    const int wm   = (wave & 3) * 32;
    const int wn   = (wave >> 2) * 64;
    const int half = lane >> 4;
    const int l16  = lane & 15;

    v8f zero = {};
    v8f acc[2][4];
#pragma unroll
    for (int i = 0; i < 2; ++i)
#pragma unroll
        for (int j = 0; j < 4; ++j) acc[i][j] = zero;

    // B staging mapping: thread -> (pixel sB_n, 32 kk values starting at sB_k0)
    const int sB_n  = t & 127;
    const int sB_k0 = (t >> 7) * 32;
    const int ph    = (n_base + sB_n) >> 5;
    const int pw    = (n_base + sB_n) & 31;

    // one 16x16x32 double-step on the current LDS tiles
    auto mma_step = [&]() {
#pragma unroll
        for (int ks = 0; ks < 2; ++ks) {
            const int kof = ks * 32;
            Frag a[2], bf[4];
#pragma unroll
            for (int i = 0; i < 2; ++i) {
                const _Float16* s = &ldsA[(wm + i * 16 + l16) * KP + kof + half * 8];
                a[i].h[0] = *(const v8h*)(s);
                a[i].h[1] = *(const v8h*)(s + 16);
            }
#pragma unroll
            for (int j = 0; j < 4; ++j) {
                const _Float16* s = &ldsB[(wn + j * 16 + l16) * KP + kof + half * 8];
                bf[j].h[0] = *(const v8h*)(s);
                bf[j].h[1] = *(const v8h*)(s + 16);
            }
#pragma unroll
            for (int i = 0; i < 2; ++i)
#pragma unroll
                for (int j = 0; j < 4; ++j)
                    acc[i][j] = __builtin_amdgcn_wmma_f32_16x16x32_f16(
                        false, a[i].v, false, bf[j].v,
                        (short)0, acc[i][j], false, false);
        }
    };

    if constexpr (PREP) {
        // -------- fast path: f16 sources, padded x, pipelined via registers --------
        const _Float16* xpb    = xpad + (size_t)b * IN_C * PIMG;
        const _Float16* xBbase = xpb + (ph + 3 - PAD) * PW38 + (pw + 3 - PAD);

        v8h      rA[4];
        _Float16 rB[32];

        auto loadA = [&](int kt) {
#pragma unroll
            for (int ps = 0; ps < 4; ++ps) {
                int m = ps * 32 + (t >> 3);
                rA[ps] = *(const v8h*)(wh + (size_t)m * Ktot + kt + (t & 7) * 8);
            }
        };
        auto storeA = [&]() {
#pragma unroll
            for (int ps = 0; ps < 4; ++ps)
                *(v8h*)&ldsA[(ps * 32 + (t >> 3)) * KP + (t & 7) * 8] = rA[ps];
        };
        auto loadB = [&](int kt) {
            const int r  = kt >> 8;                   // (kh,kw) plane of this K-tile
            const int kh = r / K, kw = r - kh * K;
            const _Float16* p = xBbase + kh * PW38 + kw
                              + (size_t)((kt & 255) + sB_k0) * PIMG;
#pragma unroll
            for (int ps = 0; ps < 32; ++ps)           // constant-imm offsets, 0 VALU
                rB[ps] = p[ps * PIMG];
        };
        auto storeB = [&]() {
#pragma unroll
            for (int q = 0; q < 16; ++q) {
                v2h p2 = { rB[2 * q], rB[2 * q + 1] };
                *(v2h*)&ldsB[sB_n * KP + sB_k0 + 2 * q] = p2;
            }
        };

        loadA(0); loadB(0);
        storeA(); storeB();
        __syncthreads();
        for (int kt = 0; kt < Ktot; kt += 64) {
            const bool more = (kt + 64) < Ktot;
            if (more) { loadA(kt + 64); loadB(kt + 64); }  // overlap with WMMA below
            mma_step();
            __syncthreads();                                // all LDS reads done
            if (more) { storeA(); storeB(); }
            __syncthreads();
        }
    } else {
        // -------- fallback: self-contained f32 sources, (c,kh,kw) K-order --------
        const float* xb = x + (size_t)b * IN_C * NPIX;
        const int sA_col = (t & 15) * 4;
        const int sA_m0  = t >> 4;
        for (int kt = 0; kt < Ktot; kt += 64) {
            __syncthreads();
            {
                const float* wp = w + (size_t)m_base * Ktot + kt + sA_col;
#pragma unroll
                for (int pass = 0; pass < 8; ++pass) {
                    int m = pass * 16 + sA_m0;
                    v4f v = *(const v4f*)(wp + (size_t)m * Ktot);
                    v4h hv = { (_Float16)v.x, (_Float16)v.y, (_Float16)v.z, (_Float16)v.w };
                    *(v4h*)&ldsA[m * KP + sA_col] = hv;
                }
            }
            for (int pass = 0; pass < 32; ++pass) {
                int kkl = pass + sB_k0;
                int kkg = kt + kkl;
                int c   = kkg / KK;
                int r   = kkg - c * KK;
                int kh  = r / K;
                int kw  = r - kh * K;
                int hh  = ph + kh - PAD;
                int ww  = pw + kw - PAD;
                float val = 0.f;
                if ((unsigned)hh < 32u && (unsigned)ww < 32u)
                    val = xb[c * NPIX + hh * 32 + ww];
                ldsB[sB_n * KP + kkl] = (_Float16)val;
            }
            __syncthreads();
            mma_step();
        }
    }

    // -------- epilogue: D vgpr r holds M = r (+8 for lane>=16); add bias --------
#pragma unroll
    for (int i = 0; i < 2; ++i) {
        const int ml0 = wm + i * 16 + half * 8;
        const int o0  = m_base + ml0;
#pragma unroll
        for (int j = 0; j < 4; ++j) {
            const int p = n_base + wn + j * 16 + l16;
            float* op = outb + (size_t)o0 * NPIX + p;
#pragma unroll
            for (int r2 = 0; r2 < 8; ++r2)
                op[r2 * NPIX] = acc[i][j][r2] + biasLds[ml0 + r2];
        }
    }
}

// ---------------- host launch ----------------
extern "C" void kernel_launch(void* const* d_in, const int* in_sizes, int n_in,
                              void* d_out, int out_size, void* d_ws, size_t ws_size,
                              hipStream_t stream) {
    const float* x   = (const float*)d_in[0];
    const int*   y   = (const int*)  d_in[1];
    const int*   arc = (const int*)  d_in[2];
    const float* wp[4] = { (const float*)d_in[3], (const float*)d_in[5],
                           (const float*)d_in[7], (const float*)d_in[9] };
    const float* ep[4] = { (const float*)d_in[4], (const float*)d_in[6],
                           (const float*)d_in[8], (const float*)d_in[10] };
    float* out = (float*)d_out;

    const size_t XPAD_ELEMS = (size_t)64 * IN_C * PIMG;          // 23,658,496 f16
    const size_t KTOT[4] = { 256, 2304, 6400, 12544 };
    size_t wseg[4], cur = XPAD_ELEMS;
    for (int i = 0; i < 4; ++i) { wseg[i] = cur; cur += KTOT[i] * OUT_C; }
    const size_t need_bytes = cur * sizeof(_Float16);            // ~58.3 MB

    dim3 grid(64 * 16), block(256);

    if (ws_size >= need_bytes) {
        _Float16* xpad = (_Float16*)d_ws;
        _Float16* wh[4];
        for (int i = 0; i < 4; ++i) wh[i] = (_Float16*)d_ws + wseg[i];

        pad_x_f16<<<dim3(64 * IN_C), block, 0, stream>>>(x, xpad);
        conv_w_f16<1><<<dim3(OUT_C), block, 0, stream>>>(wp[0], wh[0]);
        conv_w_f16<3><<<dim3(OUT_C), block, 0, stream>>>(wp[1], wh[1]);
        conv_w_f16<5><<<dim3(OUT_C), block, 0, stream>>>(wp[2], wh[2]);
        conv_w_f16<7><<<dim3(OUT_C), block, 0, stream>>>(wp[3], wh[3]);

        cond_conv_wmma<1, 0, true><<<grid, block, 0, stream>>>(x, y, arc, wp[0], ep[0], xpad, wh[0], out);
        cond_conv_wmma<3, 1, true><<<grid, block, 0, stream>>>(x, y, arc, wp[1], ep[1], xpad, wh[1], out);
        cond_conv_wmma<5, 2, true><<<grid, block, 0, stream>>>(x, y, arc, wp[2], ep[2], xpad, wh[2], out);
        cond_conv_wmma<7, 3, true><<<grid, block, 0, stream>>>(x, y, arc, wp[3], ep[3], xpad, wh[3], out);
    } else {
        cond_conv_wmma<1, 0, false><<<grid, block, 0, stream>>>(x, y, arc, wp[0], ep[0], nullptr, nullptr, out);
        cond_conv_wmma<3, 1, false><<<grid, block, 0, stream>>>(x, y, arc, wp[1], ep[1], nullptr, nullptr, out);
        cond_conv_wmma<5, 2, false><<<grid, block, 0, stream>>>(x, y, arc, wp[2], ep[2], nullptr, nullptr, out);
        cond_conv_wmma<7, 3, false><<<grid, block, 0, stream>>>(x, y, arc, wp[3], ep[3], nullptr, nullptr, out);
    }
}
